// PulseLayer_64424509440466
// MI455X (gfx1250) — matrix operand order repaired and checked by reading, not compiled, not run on hardware
//
#include <hip/hip_runtime.h>
#include <cstdint>

// ---------------------------------------------------------------------------
// PulseLayer for MI455X (gfx1250, wave32, WMMA bf16, async-LDS staging)
//   states:(8,256,1024) tokens:(8,4096,1024) -> out:(8,256,1024) f32
// One-shot bf16 pre-conversion (weights transposed to NxK), then every GEMM
// stages A and B^T tiles with global_load_async_to_lds_b128 (ASYNCcnt) and
// runs v_wmma_f32_16x16x32_bf16. Softmax reductions via DPP butterflies.
// Softmax scale (1/8) folded into the q-projection epilogue.
// ---------------------------------------------------------------------------

typedef __attribute__((ext_vector_type(16))) __bf16        bf16x16;
typedef __attribute__((ext_vector_type(8)))  float         f32x8;
typedef __attribute__((ext_vector_type(4)))  unsigned int  u32x4;

union FragB {
  bf16x16        bf;
  u32x4          q[2];
  unsigned short h[16];
};

union Pack8 {
  u32x4          q;
  unsigned short h[8];
};

__device__ __forceinline__ unsigned short f2bf(float f) {
  // round-to-nearest-even f32 -> bf16
  unsigned int u = __float_as_uint(f);
  u += 0x7FFFu + ((u >> 16) & 1u);
  return (unsigned short)(u >> 16);
}

__device__ __forceinline__ f32x8 wmma_bf16(const FragB& a, const FragB& b, f32x8 c) {
  return __builtin_amdgcn_wmma_f32_16x16x32_bf16(
      /*neg_a=*/false, a.bf, /*neg_b=*/false, b.bf,
      /*c_mod=*/(short)0, c, /*reuse_a=*/false, /*reuse_b=*/false);
}

// ---- DPP butterfly reduction within each 16-lane row (no LDS hardware) ----
template <int CTRL>
__device__ __forceinline__ float dpp_mov(float v) {
  return __int_as_float(__builtin_amdgcn_update_dpp(
      0, __float_as_int(v), CTRL, 0xF, 0xF, true));
}
__device__ __forceinline__ float red16_max(float v) {
  v = fmaxf(v, dpp_mov<0xB1>(v));    // quad_perm(1,0,3,2)  ~ xor 1
  v = fmaxf(v, dpp_mov<0x4E>(v));    // quad_perm(2,3,0,1)  ~ xor 2
  v = fmaxf(v, dpp_mov<0x141>(v));   // row_half_mirror     ~ xor 7
  v = fmaxf(v, dpp_mov<0x140>(v));   // row_mirror          ~ xor 15
  return v;
}
__device__ __forceinline__ float red16_sum(float v) {
  v += dpp_mov<0xB1>(v);
  v += dpp_mov<0x4E>(v);
  v += dpp_mov<0x141>(v);
  v += dpp_mov<0x140>(v);
  return v;
}

// ---------------------------------------------------------------------------
// One-shot converters (run once per call; off the critical compute loops)
// ---------------------------------------------------------------------------
__global__ __launch_bounds__(256)
void cvt_bf16_kernel(const float* __restrict__ src,
                     unsigned short* __restrict__ dst, int n) {
  const int i = (blockIdx.x * 256 + threadIdx.x) * 4;
  if (i < n) {
    const float4 v = *(const float4*)(src + i);
    dst[i + 0] = f2bf(v.x); dst[i + 1] = f2bf(v.y);
    dst[i + 2] = f2bf(v.z); dst[i + 3] = f2bf(v.w);
  }
}

// src: K x N f32 (row-major)  ->  dst: N x K bf16 (row-major transpose)
__global__ __launch_bounds__(256)
void cvtT_bf16_kernel(const float* __restrict__ src,
                      unsigned short* __restrict__ dst, int K, int N) {
  __shared__ float tile[32][33];
  const int k0 = blockIdx.x * 32, n0 = blockIdx.y * 32;
  const int tx = threadIdx.x & 31, ty = threadIdx.x >> 5;   // 32 x 8
#pragma unroll
  for (int i = 0; i < 4; ++i)
    tile[ty + i * 8][tx] = src[(size_t)(k0 + ty + i * 8) * N + n0 + tx];
  __syncthreads();
#pragma unroll
  for (int i = 0; i < 4; ++i)
    dst[(size_t)(n0 + ty + i * 8) * K + k0 + tx] = f2bf(tile[tx][ty + i * 8]);
}

// ---------------------------------------------------------------------------
// GEMM: C[M,N] (+)= (A[M,K] @ Wt[N,K]^T + bias[N]) * scaleOut
// A and Wt are bf16; both tiles staged by async global->LDS b128 DMA.
//   OBF16: C is bf16 (no accum); else f32 (optional accumulate)
// Block: 256 threads (8 waves). Tile BM=64, BN=128, BK=32.
// Wave (wm,wn) in 4x2 grid owns a 16x64 strip -> 4 accum tiles of 16x16.
// M%64==0, N%128==0, K%32==0 guaranteed by launch shapes.
// ---------------------------------------------------------------------------
#define GBM 64
#define GBN 128
#define GBK 32
#define ASTRIDE 40   // u16 stride: 80B row pitch (16B multiple, staggers banks)
#define BSTRIDE 40

template <int OBF16>
__global__ __launch_bounds__(256)
void gemm_kernel(const unsigned short* __restrict__ A,    // M x K bf16
                 const unsigned short* __restrict__ Wt,   // N x K bf16
                 const float* __restrict__ bias, void* __restrict__ Cp,
                 int M, int N, int K, int hasBias, int accum, float scaleOut) {
  __shared__ __align__(16) unsigned short As[GBM * ASTRIDE];   // [m][k]
  __shared__ __align__(16) unsigned short Bt[GBN * BSTRIDE];   // [n][k]

  const int tid  = threadIdx.x;
  const int lane = tid & 31, wave = tid >> 5;
  const int wm   = wave >> 1, wn = wave & 1;
  const int nl   = lane & 15, half = lane >> 4;
  const int bM   = blockIdx.y * GBM;
  const int bN   = blockIdx.x * GBN;

  f32x8 acc[4];
#pragma unroll
  for (int t = 0; t < 4; ++t) acc[t] = (f32x8){};

  const int arow = tid >> 2;          // 0..63
  const int aseg = tid & 3;           // 4 segments of 8 bf16

  for (int kb = 0; kb < K; kb += GBK) {
    __syncthreads();
    // ---- A tile 64x32: one async b128 global->LDS per thread ----
    {
      const unsigned short* src = A + (size_t)(bM + arow) * K + kb + aseg * 8;
      const unsigned ldst = (unsigned)(uintptr_t)&As[arow * ASTRIDE + aseg * 8];
      asm volatile("global_load_async_to_lds_b128 %0, %1, off"
                   :: "v"(ldst), "v"(src) : "memory");
      if (kb + GBK < K) __builtin_prefetch(src + GBK, 0, 1);
    }
    // ---- B tile 128x32 (already transposed in memory): two async b128 ----
#pragma unroll
    for (int i = 0; i < 2; ++i) {
      const int s = tid * 2 + i;             // 0..511
      const int n = s >> 2, kseg = s & 3;
      const unsigned short* src = Wt + (size_t)(bN + n) * K + kb + kseg * 8;
      const unsigned ldst = (unsigned)(uintptr_t)&Bt[n * BSTRIDE + kseg * 8];
      asm volatile("global_load_async_to_lds_b128 %0, %1, off"
                   :: "v"(ldst), "v"(src) : "memory");
      if (kb + GBK < K) __builtin_prefetch(src + GBK, 0, 1);
    }
    asm volatile("s_wait_asynccnt 0x0" ::: "memory");
    __syncthreads();

    // ---- one K=32 WMMA step per 16x16 tile ----
    // A frag (16x32 bf16): lane nl = row M; VGPR0-3 hold K=8*half..+7,
    // VGPR4-7 hold K=16+8*half..+7  -> two contiguous b128 LDS loads.
    FragB a;
    a.q[0] = *(const u32x4*)&As[(wm * 16 + nl) * ASTRIDE + 8 * half];
    a.q[1] = *(const u32x4*)&As[(wm * 16 + nl) * ASTRIDE + 16 + 8 * half];
#pragma unroll
    for (int t = 0; t < 4; ++t) {
      // B frag (32x16 bf16): lane nl = col N; K = 8*half..+7, 16+8*half..+7.
      FragB b;
      const int n = wn * 64 + t * 16 + nl;
      b.q[0] = *(const u32x4*)&Bt[n * BSTRIDE + 8 * half];
      b.q[1] = *(const u32x4*)&Bt[n * BSTRIDE + 16 + 8 * half];
      acc[t] = wmma_bf16(a, b, acc[t]);
    }
  }

  // ---- epilogue: C-layout is (M = r + 8*half, N = lane&15) per VGPR r ----
#pragma unroll
  for (int t = 0; t < 4; ++t) {
    const int n  = bN + wn * 64 + t * 16 + nl;
    const float bsv = hasBias ? bias[n] : 0.0f;
#pragma unroll
    for (int r = 0; r < 8; ++r) {
      const int m = bM + wm * 16 + r + 8 * half;
      const size_t idx = (size_t)m * N + n;
      float v = (acc[t][r] + bsv) * scaleOut;
      if constexpr (OBF16) {
        ((unsigned short*)Cp)[idx] = f2bf(v);
      } else {
        float* C = (float*)Cp;
        if (accum) v += C[idx];
        C[idx] = v;
      }
    }
  }
}

// ---------------------------------------------------------------------------
// Flash attention: per (b, h, 128 q-rows). 8 waves x 16 q-rows each.
// q/kv are bf16 (q pre-scaled by 1/8 in its projection GEMM).
// Streams S=4096 in chunks of 64 keys; K==V == kv head slice.
//   scores = q @ K^T  (inner d=64 -> two K=32 WMMA steps)
//   online softmax (DPP reductions), P -> per-wave LDS, O += P @ V
// ---------------------------------------------------------------------------
#define KSTR 72   // u16 stride for 64-wide rows (144B, 16B multiple)
#define PSTR 72

__global__ __launch_bounds__(256)
void attn_kernel(const unsigned short* __restrict__ qb,
                 const unsigned short* __restrict__ kvb,
                 const unsigned char* __restrict__ mask,
                 unsigned short* __restrict__ attb) {
  __shared__ __align__(16) unsigned short Kc[64 * KSTR];      // [s_local][d]
  __shared__ __align__(16) unsigned short Vt[64 * KSTR];      // [d][s_local]
  __shared__ __align__(16) unsigned short Pw[8][16 * PSTR];   // per-wave P

  const int tid  = threadIdx.x;
  const int lane = tid & 31, wave = tid >> 5;
  const int nl   = lane & 15, half = lane >> 4;
  const int qt = blockIdx.x, h = blockIdx.y, bb = blockIdx.z;
  const int q0 = qt * 128 + wave * 16;

  // ---- preload Q fragments: A-layout is contiguous in bf16 memory ----
  FragB qa[2];
  {
    const unsigned short* qrow =
        qb + ((size_t)(bb * 256 + q0 + nl)) * 1024 + h * 64;
#pragma unroll
    for (int kk = 0; kk < 2; ++kk) {
      qa[kk].q[0] = *(const u32x4*)(qrow + kk * 32 + 8 * half);
      qa[kk].q[1] = *(const u32x4*)(qrow + kk * 32 + 16 + 8 * half);
    }
  }

  float mrow[8], lrow[8];
  f32x8 O[4];
#pragma unroll
  for (int r = 0; r < 8; ++r) { mrow[r] = -1e30f; lrow[r] = 0.0f; }
#pragma unroll
  for (int t = 0; t < 4; ++t) O[t] = (f32x8){};

  const int crow = tid >> 2;           // 0..63  s_local row

  for (int s0 = 0; s0 < 4096; s0 += 64) {
    __syncthreads();   // all waves done with previous Kc/Vt
    // ---- stage KV chunk (64 keys x 64 dims, bf16) in two layouts ----
    {
      const unsigned short* src =
          kvb + ((size_t)(bb * 4096 + s0 + crow)) * 1024 + h * 64;
#pragma unroll
      for (int i = 0; i < 2; ++i) {
        const int seg = (tid & 3) * 2 + i;   // 0..7 (8 bf16 each)
        Pack8 p;
        p.q = *(const u32x4*)(src + seg * 8);
        *(u32x4*)&Kc[crow * KSTR + seg * 8] = p.q;   // direct b128 copy
        const int d0 = seg * 8;
#pragma unroll
        for (int j = 0; j < 8; ++j)                   // transpose for V
          Vt[(d0 + j) * KSTR + crow] = p.h[j];
      }
    }
    __syncthreads();

    // ---- scores: S[16 x 64] = q @ K^T (q pre-scaled by 1/8) ----
    f32x8 sacc[4];
#pragma unroll
    for (int t = 0; t < 4; ++t) sacc[t] = (f32x8){};
#pragma unroll
    for (int kk = 0; kk < 2; ++kk) {
#pragma unroll
      for (int t = 0; t < 4; ++t) {
        FragB b;                        // B = K^T: N = key index, K-dim = d
        const int srow = t * 16 + nl;   // key row in chunk
        b.q[0] = *(const u32x4*)&Kc[srow * KSTR + kk * 32 + 8 * half];
        b.q[1] = *(const u32x4*)&Kc[srow * KSTR + kk * 32 + 16 + 8 * half];
        sacc[t] = wmma_bf16(qa[kk], b, sacc[t]);
      }
    }

    // ---- attention mask (bool bytes; -1e30 keeps exp() finite) ----
#pragma unroll
    for (int t = 0; t < 4; ++t) {
      if (!mask[(size_t)bb * 4096 + s0 + t * 16 + nl]) {
#pragma unroll
        for (int r = 0; r < 8; ++r) sacc[t][r] = -1e30f;
      }
    }

    // ---- online softmax; row (r + 8*half) spans 16 lanes x 4 tiles ----
    float alpha[8];
#pragma unroll
    for (int r = 0; r < 8; ++r) {
      float mx = fmaxf(fmaxf(sacc[0][r], sacc[1][r]),
                       fmaxf(sacc[2][r], sacc[3][r]));
      mx = red16_max(mx);                       // DPP butterfly, 16 lanes
      const float mnew = fmaxf(mrow[r], mx);
      alpha[r] = __expf(mrow[r] - mnew);
      float ls = 0.0f;
#pragma unroll
      for (int t = 0; t < 4; ++t) {
        const float p = __expf(sacc[t][r] - mnew);
        sacc[t][r] = p;
        ls += p;
      }
      ls = red16_sum(ls);                       // DPP butterfly, 16 lanes
      lrow[r] = lrow[r] * alpha[r] + ls;
      mrow[r] = mnew;
    }
#pragma unroll
    for (int t = 0; t < 4; ++t)
#pragma unroll
      for (int r = 0; r < 8; ++r) O[t][r] *= alpha[r];

    // ---- P (C-layout) -> per-wave LDS so it can re-enter as A-matrix ----
    unsigned short* P = Pw[wave];
#pragma unroll
    for (int t = 0; t < 4; ++t)
#pragma unroll
      for (int r = 0; r < 8; ++r)
        P[(r + 8 * half) * PSTR + t * 16 + nl] = f2bf(sacc[t][r]);
    asm volatile("s_wait_dscnt 0x0" ::: "memory");  // wave-local LDS RAW fence

    // ---- O[16 x 64] += P @ V  (inner dim = s_local, two K=32 steps) ----
#pragma unroll
    for (int kk = 0; kk < 2; ++kk) {
      FragB a;
      a.q[0] = *(const u32x4*)&P[nl * PSTR + kk * 32 + 8 * half];
      a.q[1] = *(const u32x4*)&P[nl * PSTR + kk * 32 + 16 + 8 * half];
#pragma unroll
      for (int t = 0; t < 4; ++t) {
        FragB b;                        // B = V: N = d, K-dim = s_local
        const int drow = t * 16 + nl;
        b.q[0] = *(const u32x4*)&Vt[drow * KSTR + kk * 32 + 8 * half];
        b.q[1] = *(const u32x4*)&Vt[drow * KSTR + kk * 32 + 16 + 8 * half];
        O[t] = wmma_bf16(a, b, O[t]);
      }
    }
  }

  // ---- normalize and write attended (bf16, pre-W_out) ----
#pragma unroll
  for (int r = 0; r < 8; ++r) {
    const float inv = 1.0f / lrow[r];
    const int row = q0 + r + 8 * half;
#pragma unroll
    for (int t = 0; t < 4; ++t)
      attb[((size_t)(bb * 256 + row)) * 1024 + h * 64 + t * 16 + nl] =
          f2bf(O[t][r] * inv);
  }
}

// ---------------------------------------------------------------------------
// x = states + upd; LayerNorm(x) * gamma + beta. One block per row of 1024.
// ---------------------------------------------------------------------------
__global__ __launch_bounds__(256)
void add_ln_kernel(const float* __restrict__ states, const float* __restrict__ upd,
                   const float* __restrict__ gamma, const float* __restrict__ beta,
                   float* __restrict__ out) {
  const int row  = blockIdx.x;
  const int tid  = threadIdx.x;
  const int lane = tid & 31, wave = tid >> 5;
  __shared__ float rs[8], rq[8];

  const float* ps = states + (size_t)row * 1024;
  const float* pu = upd    + (size_t)row * 1024;

  float x[4], s = 0.0f, sq = 0.0f;
#pragma unroll
  for (int i = 0; i < 4; ++i) {
    const int j = tid + i * 256;
    const float v = ps[j] + pu[j];
    x[i] = v; s += v; sq += v * v;
  }
#pragma unroll
  for (int off = 16; off; off >>= 1) {
    s  += __shfl_xor(s,  off, 32);
    sq += __shfl_xor(sq, off, 32);
  }
  if (lane == 0) { rs[wave] = s; rq[wave] = sq; }
  __syncthreads();
  float ts = 0.0f, tq = 0.0f;
#pragma unroll
  for (int w = 0; w < 8; ++w) { ts += rs[w]; tq += rq[w]; }
  const float mu   = ts * (1.0f / 1024.0f);
  const float var  = tq * (1.0f / 1024.0f) - mu * mu;
  const float rstd = rsqrtf(var + 1e-5f);
#pragma unroll
  for (int i = 0; i < 4; ++i) {
    const int j = tid + i * 256;
    out[(size_t)row * 1024 + j] = (x[i] - mu) * rstd * gamma[j] + beta[j];
  }
}

// ---------------------------------------------------------------------------
// Launcher. Input order: states, tokens, attention_mask, W_state, b_state,
// W_token, b_token, W_out, b_out, W_upd, b_upd, ln_gamma, ln_beta.
// ---------------------------------------------------------------------------
extern "C" void kernel_launch(void* const* d_in, const int* in_sizes, int n_in,
                              void* d_out, int out_size, void* d_ws, size_t ws_size,
                              hipStream_t stream) {
  const float* states  = (const float*)d_in[0];
  const float* tokens  = (const float*)d_in[1];
  const unsigned char* mask = (const unsigned char*)d_in[2];
  const float* W_state = (const float*)d_in[3];
  const float* b_state = (const float*)d_in[4];
  const float* W_token = (const float*)d_in[5];
  const float* b_token = (const float*)d_in[6];
  const float* W_out   = (const float*)d_in[7];
  const float* b_out   = (const float*)d_in[8];
  const float* W_upd   = (const float*)d_in[9];
  const float* b_upd   = (const float*)d_in[10];
  const float* gamma   = (const float*)d_in[11];
  const float* beta    = (const float*)d_in[12];
  float* out = (float*)d_out;

  const size_t MQ = (size_t)2048 * 1024;     // states-shaped (elems)
  const size_t MT = (size_t)32768 * 1024;    // tokens-shaped (elems)
  const size_t MW = (size_t)1024 * 1024;     // one 1024x1024 weight (elems)

  unsigned short* st_bf  = (unsigned short*)d_ws;        // bf16 states
  unsigned short* tok_bf = st_bf  + MQ;                  // bf16 tokens
  unsigned short* q_bf   = tok_bf + MT;                  // q (pre-scaled)
  unsigned short* kv_bf  = q_bf   + MQ;                  // kv
  unsigned short* att_bf = kv_bf  + MT;                  // attended
  unsigned short* ato_bf = att_bf + MQ;                  // attended @ W_out
  unsigned short* Wt_st  = ato_bf + MQ;                  // W_state^T
  unsigned short* Wt_tk  = Wt_st  + MW;                  // W_token^T
  unsigned short* Wt_ot  = Wt_tk  + MW;                  // W_out^T
  unsigned short* Wt_u0  = Wt_ot  + MW;                  // W_upd[0:1024]^T
  unsigned short* Wt_u1  = Wt_u0  + MW;                  // W_upd[1024:2048]^T
  float*          upd_f  = (float*)(Wt_u1 + MW);         // f32 update

  const dim3 blk(256);
  const dim3 wgrid(32, 32);   // 1024x1024 transpose-convert

  // ---- one-shot bf16 conversions ----
  cvt_bf16_kernel<<<dim3(MQ / 1024), blk, 0, stream>>>(states, st_bf, (int)MQ);
  cvt_bf16_kernel<<<dim3(MT / 1024), blk, 0, stream>>>(tokens, tok_bf, (int)MT);
  cvtT_bf16_kernel<<<wgrid, blk, 0, stream>>>(W_state, Wt_st, 1024, 1024);
  cvtT_bf16_kernel<<<wgrid, blk, 0, stream>>>(W_token, Wt_tk, 1024, 1024);
  cvtT_bf16_kernel<<<wgrid, blk, 0, stream>>>(W_out,   Wt_ot, 1024, 1024);
  cvtT_bf16_kernel<<<wgrid, blk, 0, stream>>>(W_upd,            Wt_u0, 1024, 1024);
  cvtT_bf16_kernel<<<wgrid, blk, 0, stream>>>(W_upd + MW,       Wt_u1, 1024, 1024);

  // ---- q = (states @ W_state + b_state) / sqrt(64)  -> bf16 ----
  gemm_kernel<1><<<dim3(1024 / GBN, 2048 / GBM), blk, 0, stream>>>(
      st_bf, Wt_st, b_state, q_bf, 2048, 1024, 1024, 1, 0, 0.125f);
  // ---- kv = tokens @ W_token + b_token  -> bf16  (dominant GEMM) ----
  gemm_kernel<1><<<dim3(1024 / GBN, 32768 / GBM), blk, 0, stream>>>(
      tok_bf, Wt_tk, b_token, kv_bf, 32768, 1024, 1024, 1, 0, 1.0f);
  // ---- attention -> att_bf ----
  attn_kernel<<<dim3(2, 16, 8), blk, 0, stream>>>(q_bf, kv_bf, mask, att_bf);
  // ---- att_out = att @ W_out + b_out -> bf16 ----
  gemm_kernel<1><<<dim3(1024 / GBN, 2048 / GBM), blk, 0, stream>>>(
      att_bf, Wt_ot, b_out, ato_bf, 2048, 1024, 1024, 1, 0, 1.0f);
  // ---- upd = states @ W_upd[:1024] + b_upd ; upd += att_out @ W_upd[1024:] ----
  gemm_kernel<0><<<dim3(1024 / GBN, 2048 / GBM), blk, 0, stream>>>(
      st_bf, Wt_u0, b_upd, upd_f, 2048, 1024, 1024, 1, 0, 1.0f);
  gemm_kernel<0><<<dim3(1024 / GBN, 2048 / GBM), blk, 0, stream>>>(
      ato_bf, Wt_u1, nullptr, upd_f, 2048, 1024, 1024, 0, 1, 1.0f);
  // ---- out = LayerNorm(states + upd) * gamma + beta ----
  add_ln_kernel<<<dim3(2048), blk, 0, stream>>>(states, upd_f, gamma, beta, out);
}
